// FusionGNN_49392123904094
// MI455X (gfx1250) — matrix operand order repaired and checked by reading, not compiled, not run on hardware
//
#include <hip/hip_runtime.h>
#include <hip/hip_bf16.h>
#include <stdint.h>

// ---------------- CDNA5 WMMA types ----------------
typedef __attribute__((ext_vector_type(16))) __bf16         v16bf;
typedef __attribute__((ext_vector_type(8)))  float          v8f;
typedef __attribute__((ext_vector_type(8)))  unsigned short v8us;
typedef __attribute__((ext_vector_type(16))) unsigned short v16us;

union BFrag { v16us us; v16bf bf; };

__device__ __forceinline__ unsigned short f2bf(float f) {
  union { float f; unsigned u; } c; c.f = f;
  unsigned r = c.u + 0x7FFFu + ((c.u >> 16) & 1u);   // round-to-nearest-even
  return (unsigned short)(r >> 16);
}

// ---------------- trivial utility kernels ----------------
__global__ void k_zero(float* __restrict__ p, int n) {
  int i = blockIdx.x * blockDim.x + threadIdx.x;
  if (i < n) p[i] = 0.0f;
}

__global__ void k_count(const int* __restrict__ dst, float* __restrict__ cnt, int E) {
  int e = blockIdx.x * blockDim.x + threadIdx.x;
  if (e < E) atomicAdd(&cnt[dst[e]], 1.0f);
}

// agg[dst[e]][c] += h[src[e]][c]   (one thread per edge-channel)
__global__ void k_scatter(const float* __restrict__ h, const int* __restrict__ src,
                          const int* __restrict__ dst, float* __restrict__ agg,
                          int C, int total) {
  int i = blockIdx.x * blockDim.x + threadIdx.x;
  if (i >= total) return;
  int e = i / C, c = i - e * C;
  atomicAdd(&agg[(size_t)dst[e] * C + c], h[(size_t)src[e] * C + c]);
}

// pack [A | B | 0-pad] row-wise into bf16 with row length Kpad
__global__ void k_pack_w(const float* __restrict__ A, int Ka,
                         const float* __restrict__ B, int Kb,
                         unsigned short* __restrict__ Wb, int Kpad, int total) {
  int i = blockIdx.x * blockDim.x + threadIdx.x;
  if (i >= total) return;
  int n = i / Kpad, k = i - n * Kpad;
  float v = 0.0f;
  if (k < Ka)                 v = A[(size_t)n * Ka + k];
  else if (B && k < Ka + Kb)  v = B[(size_t)n * Kb + (k - Ka)];
  Wb[i] = f2bf(v);
}

// ---------------- fused SAGEConv GEMM ----------------
// out[node] = relu( [agg[node]/max(cnt,1) , self[node]] @ Wb.T + bias )
// Block: 256 threads = 8 wave32; block covers 16 nodes x 128 cols;
// each wave computes one 16x16 tile via v_wmma_f32_16x16x32_bf16.
__global__ __launch_bounds__(256) void k_sage_gemm(
    const float* __restrict__ agg, const float* __restrict__ cnt,
    const float* __restrict__ self, const unsigned short* __restrict__ Wb,
    const float* __restrict__ bias, int C1, int C2, int Kpad,
    float* __restrict__ outF, unsigned short* __restrict__ outB)
{
  __shared__ unsigned short lds_a[16 * 256];   // Kpad <= 256 here
  __shared__ float lds_inv[16];

  const int tid = threadIdx.x;
  const int nodeBase = blockIdx.x * 16;

  if (tid < 16) lds_inv[tid] = 1.0f / fmaxf(cnt[nodeBase + tid], 1.0f);
  __syncthreads();

  // Stage A tile (16 rows x Kpad) as bf16: [mean | self | 0-pad]
  for (int i = tid; i < 16 * Kpad; i += 256) {
    int row = i / Kpad, k = i - row * Kpad;
    size_t node = (size_t)(nodeBase + row);
    float v = 0.0f;
    if (k < C1)            v = agg[node * C1 + k] * lds_inv[row];
    else if (k < C1 + C2)  v = self[node * C2 + (k - C1)];
    lds_a[row * Kpad + k] = f2bf(v);
  }
  __syncthreads();

  const int lane  = tid & 31;
  const int wave  = tid >> 5;
  const int mrow  = lane & 15;
  const int akoff = (lane & 16) ? 8 : 0;    // ISA 16-bit A layout
  const int bkoff = (lane & 16) ? 8 : 0;    // ISA 16-bit B layout
  const int ncol  = wave * 16 + (lane & 15);

  v8f acc = {};
  for (int kk = 0; kk < Kpad; kk += 32) {
    BFrag a, b;
    const v8us* ap0 = (const v8us*)&lds_a[mrow * Kpad + kk + akoff];       // K: kk+akoff..+7
    const v8us* ap1 = (const v8us*)&lds_a[mrow * Kpad + kk + 16 + akoff];  // K: +16..+23
    v8us alo = ap0[0], ahi = ap1[0];
    const v8us* bp = (const v8us*)&Wb[(size_t)ncol * Kpad + kk + bkoff];   // K: kk+bkoff..+7, +16..+23
    v8us blo = bp[0], bhi = bp[2];
#pragma unroll
    for (int j = 0; j < 8; ++j) {
      a.us[j] = alo[j]; a.us[8 + j] = ahi[j];
      b.us[j] = blo[j]; b.us[8 + j] = bhi[j];
    }
    acc = __builtin_amdgcn_wmma_f32_16x16x32_bf16(false, a.bf, false, b.bf,
                                                  (short)0, acc, false, false);
  }

  const float bv = bias[ncol];
  const int mhalf = (lane & 16) ? 8 : 0;     // C/D layout: VGPR r -> M=r or r+8
#pragma unroll
  for (int r = 0; r < 8; ++r) {
    float v = fmaxf(acc[r] + bv, 0.0f);
    size_t o = (size_t)(nodeBase + mhalf + r) * 128 + ncol;
    if (outF) outF[o] = v;
    if (outB) outB[o] = f2bf(v);
  }
}

// ---------------- fused edge MLP ----------------
// per edge: ef = [h2[src], h2[dst], eattr] (K=262, pad 288)
// hidden = relu(ef @ W1.T + b1); out = hidden . w2 + b2  (layer-2 fused via LDS reduce)
__global__ __launch_bounds__(256) void k_edge_mlp(
    const unsigned short* __restrict__ h2,
    const int* __restrict__ src, const int* __restrict__ dst,
    const float* __restrict__ eattr,
    const unsigned short* __restrict__ W1b,
    const float* __restrict__ b1,
    const float* __restrict__ w2, const float* __restrict__ b2,
    float* __restrict__ out)
{
  const int KP = 288;
  __shared__ unsigned short lds_a[16 * 288];
  __shared__ float lds_out[16];

  const int tid = threadIdx.x;
  const int eBase = blockIdx.x * 16;
  if (tid < 16) lds_out[tid] = 0.0f;

  for (int i = tid; i < 16 * KP; i += 256) {
    int row = i / KP, k = i - row * KP;
    int e = eBase + row;
    int s = src[e], d = dst[e];
    unsigned short v = 0;
    if (k < 128)       v = h2[(size_t)s * 128 + k];
    else if (k < 256)  v = h2[(size_t)d * 128 + (k - 128)];
    else if (k < 262)  v = f2bf(eattr[(size_t)e * 6 + (k - 256)]);
    lds_a[row * KP + k] = v;
  }
  __syncthreads();

  const int lane  = tid & 31;
  const int wave  = tid >> 5;
  const int mrow  = lane & 15;
  const int akoff = (lane & 16) ? 8 : 0;
  const int bkoff = (lane & 16) ? 8 : 0;
  const int ncol  = wave * 16 + (lane & 15);

  v8f acc = {};
#pragma unroll
  for (int kk = 0; kk < KP; kk += 32) {
    BFrag a, b;
    const v8us* ap0 = (const v8us*)&lds_a[mrow * KP + kk + akoff];
    const v8us* ap1 = (const v8us*)&lds_a[mrow * KP + kk + 16 + akoff];
    v8us alo = ap0[0], ahi = ap1[0];
    const v8us* bp = (const v8us*)&W1b[(size_t)ncol * KP + kk + bkoff];
    v8us blo = bp[0], bhi = bp[2];
#pragma unroll
    for (int j = 0; j < 8; ++j) {
      a.us[j] = alo[j]; a.us[8 + j] = ahi[j];
      b.us[j] = blo[j]; b.us[8 + j] = bhi[j];
    }
    acc = __builtin_amdgcn_wmma_f32_16x16x32_bf16(false, a.bf, false, b.bf,
                                                  (short)0, acc, false, false);
  }

  const float bv = b1[ncol];
  const float wv = w2[ncol];
  const int mhalf = (lane & 16) ? 8 : 0;
#pragma unroll
  for (int r = 0; r < 8; ++r) {
    float h = fmaxf(acc[r] + bv, 0.0f);      // ReLU on hidden, then layer-2 dot
    atomicAdd(&lds_out[mhalf + r], h * wv);
  }
  __syncthreads();
  if (tid < 16) out[eBase + tid] = lds_out[tid] + b2[0];
}

// ---------------- host launcher ----------------
extern "C" void kernel_launch(void* const* d_in, const int* in_sizes, int n_in,
                              void* d_out, int out_size, void* d_ws, size_t ws_size,
                              hipStream_t stream) {
  (void)in_sizes; (void)n_in; (void)out_size; (void)ws_size;
  const int N = 100000, E = 600000, CIN = 42, H = 128;

  const float* x      = (const float*)d_in[0];
  const int*   ei     = (const int*)d_in[1];       // jax default x64-off -> int32
  const int*   src    = ei;
  const int*   dst    = ei + E;
  const float* eattr  = (const float*)d_in[2];
  const float* enc_Wl = (const float*)d_in[3];
  const float* enc_bl = (const float*)d_in[4];
  const float* enc_Wr = (const float*)d_in[5];
  const float* c1_Wl  = (const float*)d_in[6];
  const float* c1_bl  = (const float*)d_in[7];
  const float* c1_Wr  = (const float*)d_in[8];
  const float* mW1    = (const float*)d_in[9];
  const float* mb1    = (const float*)d_in[10];
  const float* mW2    = (const float*)d_in[11];
  const float* mb2    = (const float*)d_in[12];
  float* out = (float*)d_out;

  // workspace carve-up (~145 MB total)
  char* ws = (char*)d_ws;
  size_t off = 0;
  auto take = [&](size_t bytes) -> char* {
    char* p = ws + off;
    off = (off + bytes + 255) & ~(size_t)255;
    return p;
  };
  float*          agg1 = (float*)take((size_t)N * CIN * 4);
  float*          cntb = (float*)take((size_t)N * 4);
  float*          h1   = (float*)take((size_t)N * H * 4);
  float*          agg2 = (float*)take((size_t)N * H * 4);
  unsigned short* h2   = (unsigned short*)take((size_t)N * H * 2);
  unsigned short* Wenc = (unsigned short*)take((size_t)H * 96 * 2);
  unsigned short* Wc1  = (unsigned short*)take((size_t)H * 256 * 2);
  unsigned short* W1b  = (unsigned short*)take((size_t)H * 288 * 2);

  auto nb = [](int n, int b) { return (n + b - 1) / b; };

  // init accumulators (every call -> deterministic)
  k_zero<<<nb(N * CIN, 256), 256, 0, stream>>>(agg1, N * CIN);
  k_zero<<<nb(N, 256), 256, 0, stream>>>(cntb, N);
  k_zero<<<nb(N * H, 256), 256, 0, stream>>>(agg2, N * H);

  // degree + stage-1 scatter
  k_count<<<nb(E, 256), 256, 0, stream>>>(dst, cntb, E);
  k_scatter<<<nb(E * CIN, 256), 256, 0, stream>>>(x, src, dst, agg1, CIN, E * CIN);

  // pack bf16 weights: [Wl | Wr | pad]
  k_pack_w<<<nb(H * 96, 256), 256, 0, stream>>>(enc_Wl, CIN, enc_Wr, CIN, Wenc, 96, H * 96);
  k_pack_w<<<nb(H * 256, 256), 256, 0, stream>>>(c1_Wl, H, c1_Wr, H, Wc1, 256, H * 256);
  k_pack_w<<<nb(H * 288, 256), 256, 0, stream>>>(mW1, 262, nullptr, 0, W1b, 288, H * 288);

  // stage 1: h1 = relu([mean(x) | x] @ Wenc.T + bl)   (fp32 out, feeds scatter)
  k_sage_gemm<<<N / 16, 256, 0, stream>>>(agg1, cntb, x, Wenc, enc_bl, CIN, CIN, 96, h1, nullptr);

  // stage 2 scatter + GEMM: h2 = relu([mean(h1) | h1] @ Wc1.T + bl)  (bf16 out)
  k_scatter<<<nb(E * H, 256), 256, 0, stream>>>(h1, src, dst, agg2, H, E * H);
  k_sage_gemm<<<N / 16, 256, 0, stream>>>(agg2, cntb, h1, Wc1, c1_bl, H, H, 256, nullptr, h2);

  // stage 3: fused edge MLP (gather + GEMM + ReLU + dot w2 + b2)
  k_edge_mlp<<<E / 16, 256, 0, stream>>>(h2, src, dst, eattr, W1b, mb1, mW2, mb2, out);
}
